// CustomNetwork_54039278518555
// MI455X (gfx1250) — compile-verified, matrix operand order checked by reading, not run on hardware
//
#include <hip/hip_runtime.h>
#include <hip/hip_bf16.h>
#include <stdint.h>

// ---------------------------------------------------------------------------
// Problem constants (from reference)
// ---------------------------------------------------------------------------
#define M_TOTAL  32768      // rows (time steps)
#define F_IN     73         // feature width
#define H_FNN    512        // FNN hidden
#define D_FNN    256        // FNN out
#define H_RNN    128        // LSTM hidden
#define G4       512        // 4*H_RNN gate width
#define OUT_COLS 384        // 256 + 128
#define OUT_TOT  (M_TOTAL * OUT_COLS)

// GEMM tiling: 256 threads = 8 wave32, arranged 4(M) x 2(N).
// Each wave owns a 16x32 C strip (two 16x16 WMMA subtiles sharing one A
// fragment) -> workgroup tile = 64(M) x 64(N), K-slab = 32.
#define MT 64
#define NTW 64              // workgroup N tile
#define KT 32

typedef __attribute__((ext_vector_type(16))) __bf16 v16bf;
typedef __attribute__((ext_vector_type(8)))  float  v8f;

#define GLOBAL_AS __attribute__((address_space(1)))
#define LDS_AS    __attribute__((address_space(3)))

// exact pointee type of the async-LDS builtins (per hipcc diagnostic)
typedef int v4i_vec __attribute__((vector_size(4 * sizeof(int))));
typedef GLOBAL_AS v4i_vec* g_v4i_p;
typedef LDS_AS    v4i_vec* l_v4i_p;

#if defined(__has_builtin)
#  if __has_builtin(__builtin_amdgcn_global_load_async_to_lds_b128)
#    define HAVE_ASYNC_LDS 1
#  endif
#endif
#ifndef HAVE_ASYNC_LDS
#  define HAVE_ASYNC_LDS 0
#endif

static __device__ __forceinline__ void wait_async0() {
#if HAVE_ASYNC_LDS
#  if __has_builtin(__builtin_amdgcn_s_wait_asynccnt)
    __builtin_amdgcn_s_wait_asynccnt(0);
#  else
    asm volatile("s_wait_asynccnt 0x0" ::: "memory");
#  endif
#endif
}

__constant__ int c_dset[25] = {0,49,50,51,52,53,54,55,56,57,58,59,60,
                               61,62,63,64,65,66,67,68,69,70,71,72};

// fp32 -> bf16 (RNE), packed pair -> one dword
static __device__ __forceinline__ unsigned int f2bf(float f) {
    unsigned int u = __float_as_uint(f);
    u = u + 0x7FFFu + ((u >> 16) & 1u);
    return u >> 16;
}
static __device__ __forceinline__ unsigned int pack2(float a, float b) {
    return f2bf(a) | (f2bf(b) << 16);
}

// Build one bf16 WMMA operand fragment from an LDS tile stored as
// [row][k] (row = M for A, N for B; 32 bf16 per row, 4B-aligned).
// ISA 16-bit 16x32 layout: lane group g = lane>>4 selects K halves
// {0..7,16..23} (g=0) / {8..15,24..31} (g=1); VGPR q holds K pair
// (q<4 ? 8g+2q : 16+8g+2(q-4)) -- consecutive in K, one dword read each.
static __device__ __forceinline__ v16bf load_frag(const unsigned short* lds,
                                                  int row, int g) {
    union { v16bf v; unsigned int u[8]; } f;
    const unsigned int* p = (const unsigned int*)(lds + row * KT);
#pragma unroll
    for (int q = 0; q < 8; ++q) {
        int kb = (q < 4) ? (8 * g + 2 * q) : (16 + 8 * g + 2 * (q - 4));
        f.u[q] = p[kb >> 1];
    }
    return f.v;
}

#define WMMA_BF16(A, B, C) \
    __builtin_amdgcn_wmma_f32_16x16x32_bf16(false, (A), false, (B), (short)0, (C), false, false)

// ---------------------------------------------------------------------------
// GEMM1: Hbf(32768x512) = relu(features(32768x73) @ W1^T + b1), stored bf16
// ---------------------------------------------------------------------------
__global__ __launch_bounds__(256)
void gemm1_kernel(const float* __restrict__ A, const float* __restrict__ W1,
                  const float* __restrict__ b1, unsigned short* __restrict__ Hout) {
    __shared__ __attribute__((aligned(16))) unsigned short As[MT * KT];
    __shared__ __attribute__((aligned(16))) unsigned short Bs[NTW * KT];
    const int tid  = threadIdx.x;
    const int lane = tid & 31, wave = tid >> 5;
    const int wm = wave >> 1, wn = wave & 1;
    const int m0 = blockIdx.y * MT, n0 = blockIdx.x * NTW;
    const int g = lane >> 4;
    v8f acc0 = {0.f,0.f,0.f,0.f,0.f,0.f,0.f,0.f};
    v8f acc1 = {0.f,0.f,0.f,0.f,0.f,0.f,0.f,0.f};

    for (int k0 = 0; k0 < 96; k0 += KT) {          // K = 73 padded to 96
        // A tile 64x32: 1024 bf16 pairs, 4 per thread, branchless edge.
#pragma unroll
        for (int e = 0; e < 4; ++e) {
            int pidx = e * 256 + tid;
            int r = pidx >> 4, kp = (pidx & 15) * 2;
            int kk = k0 + kp;
            const float* row = A + (size_t)(m0 + r) * F_IN;
            float a0 = row[kk     < F_IN ? kk     : F_IN - 1];
            float a1 = row[kk + 1 < F_IN ? kk + 1 : F_IN - 1];
            a0 = (kk     < F_IN) ? a0 : 0.f;
            a1 = (kk + 1 < F_IN) ? a1 : 0.f;
            ((unsigned int*)As)[r * (KT / 2) + (kp >> 1)] = pack2(a0, a1);
        }
        // B tile 64(N)x32(K): 1024 pairs, 4 per thread.
#pragma unroll
        for (int e = 0; e < 4; ++e) {
            int pidx = e * 256 + tid;
            int n = pidx >> 4, kp = (pidx & 15) * 2;
            int kk = k0 + kp;
            const float* row = W1 + (size_t)(n0 + n) * F_IN;
            float b0 = row[kk     < F_IN ? kk     : F_IN - 1];
            float b1v = row[kk + 1 < F_IN ? kk + 1 : F_IN - 1];
            b0  = (kk     < F_IN) ? b0  : 0.f;
            b1v = (kk + 1 < F_IN) ? b1v : 0.f;
            ((unsigned int*)Bs)[n * (KT / 2) + (kp >> 1)] = pack2(b0, b1v);
        }
        __syncthreads();
        v16bf af  = load_frag(As, wm * 16 + (lane & 15), g);
        v16bf bf0 = load_frag(Bs, wn * 32 + (lane & 15), g);
        v16bf bf1 = load_frag(Bs, wn * 32 + 16 + (lane & 15), g);
        acc0 = WMMA_BF16(af, bf0, acc0);
        acc1 = WMMA_BF16(af, bf1, acc1);
        __syncthreads();
    }
    // C layout: VGPR r of lane l -> (M = r + 8*(l>>4), N = l&15)
    int nA = n0 + wn * 32 + (lane & 15);
    int nB = nA + 16;
    float biasA = b1[nA], biasB = b1[nB];
#pragma unroll
    for (int r = 0; r < 8; ++r) {
        int m = m0 + wm * 16 + r + 8 * g;
        float vA = acc0[r] + biasA; vA = vA > 0.f ? vA : 0.f;
        float vB = acc1[r] + biasB; vB = vB > 0.f ? vB : 0.f;
        Hout[(size_t)m * H_FNN + nA] = (unsigned short)f2bf(vA);
        Hout[(size_t)m * H_FNN + nB] = (unsigned short)f2bf(vB);
    }
}

// ---------------------------------------------------------------------------
// GEMM2: out[:, :256] = relu(Hbf(32768x512) @ W2^T + b2)  (both tuple copies)
// A operand is already bf16 in memory -> async global->LDS copy (ASYNCcnt).
// ---------------------------------------------------------------------------
__global__ __launch_bounds__(256)
void gemm2_kernel(const unsigned short* __restrict__ Hbf, const float* __restrict__ W2,
                  const float* __restrict__ b2, float* __restrict__ out) {
    __shared__ __attribute__((aligned(16))) unsigned short As[MT * KT];
    __shared__ __attribute__((aligned(16))) unsigned short Bs[NTW * KT];
    const int tid  = threadIdx.x;
    const int lane = tid & 31, wave = tid >> 5;
    const int wm = wave >> 1, wn = wave & 1;
    const int m0 = blockIdx.y * MT, n0 = blockIdx.x * NTW;
    const int g = lane >> 4;
    v8f acc0 = {0.f,0.f,0.f,0.f,0.f,0.f,0.f,0.f};
    v8f acc1 = {0.f,0.f,0.f,0.f,0.f,0.f,0.f,0.f};

    // A tile: 64 rows x 64B = 4 KB = 256 lanes x 16B.
    const int ar    = tid >> 2;          // row 0..63
    const int aoff  = (tid & 3) * 8;     // ushort offset within row, 16B step

    for (int k0 = 0; k0 < H_FNN; k0 += KT) {       // K = 512
        const unsigned short* gA = Hbf + (size_t)(m0 + ar) * H_FNN + k0 + aoff;
#if HAVE_ASYNC_LDS
        __builtin_amdgcn_global_load_async_to_lds_b128(
            (g_v4i_p)(void*)gA,
            (l_v4i_p)(void*)(As + ar * KT + aoff), 0, 0);
#else
        *(uint4*)(As + ar * KT + aoff) = *(const uint4*)gA;
#endif
        // prefetch next slab of W2 (lowers to global_prefetch_b8)
        if (k0 + KT < H_FNN)
            __builtin_prefetch(W2 + (size_t)(n0 + ar) * H_FNN + k0 + KT, 0, 1);
        // B tile 64(N)x32(K): fp32 -> packed bf16
#pragma unroll
        for (int e = 0; e < 4; ++e) {
            int pidx = e * 256 + tid;
            int n = pidx >> 4, kp = (pidx & 15) * 2;
            const float* row = W2 + (size_t)(n0 + n) * H_FNN + k0;
            ((unsigned int*)Bs)[n * (KT / 2) + (kp >> 1)] = pack2(row[kp], row[kp + 1]);
        }
        wait_async0();
        __syncthreads();
        v16bf af  = load_frag(As, wm * 16 + (lane & 15), g);
        v16bf bf0 = load_frag(Bs, wn * 32 + (lane & 15), g);
        v16bf bf1 = load_frag(Bs, wn * 32 + 16 + (lane & 15), g);
        acc0 = WMMA_BF16(af, bf0, acc0);
        acc1 = WMMA_BF16(af, bf1, acc1);
        __syncthreads();
    }
    int nA = n0 + wn * 32 + (lane & 15);
    int nB = nA + 16;
    float biasA = b2[nA], biasB = b2[nB];
#pragma unroll
    for (int r = 0; r < 8; ++r) {
        int m = m0 + wm * 16 + r + 8 * g;
        float vA = acc0[r] + biasA; vA = vA > 0.f ? vA : 0.f;
        float vB = acc1[r] + biasB; vB = vB > 0.f ? vB : 0.f;
        out[(size_t)m * OUT_COLS + nA] = vA;
        out[(size_t)m * OUT_COLS + nB] = vB;
        out[(size_t)OUT_TOT + (size_t)m * OUT_COLS + nA] = vA;
        out[(size_t)OUT_TOT + (size_t)m * OUT_COLS + nB] = vB;
    }
}

// ---------------------------------------------------------------------------
// GEMM3: Xg(32768x512) = features[:, D_SET] @ W_ih^T + b_ih + b_hh   (fp32)
// K = 25 -> single 32-wide slab with column gather.
// ---------------------------------------------------------------------------
__global__ __launch_bounds__(256)
void gemm3_kernel(const float* __restrict__ A, const float* __restrict__ Wih,
                  const float* __restrict__ bih, const float* __restrict__ bhh,
                  float* __restrict__ Xg) {
    __shared__ __attribute__((aligned(16))) unsigned short As[MT * KT];
    __shared__ __attribute__((aligned(16))) unsigned short Bs[NTW * KT];
    const int tid  = threadIdx.x;
    const int lane = tid & 31, wave = tid >> 5;
    const int wm = wave >> 1, wn = wave & 1;
    const int m0 = blockIdx.y * MT, n0 = blockIdx.x * NTW;
    const int g = lane >> 4;

#pragma unroll
    for (int e = 0; e < 4; ++e) {
        int pidx = e * 256 + tid;
        int r = pidx >> 4, kp = (pidx & 15) * 2;
        const float* row = A + (size_t)(m0 + r) * F_IN;
        float a0 = row[c_dset[kp     < 25 ? kp     : 24]];
        float a1 = row[c_dset[kp + 1 < 25 ? kp + 1 : 24]];
        a0 = (kp     < 25) ? a0 : 0.f;
        a1 = (kp + 1 < 25) ? a1 : 0.f;
        ((unsigned int*)As)[r * (KT / 2) + (kp >> 1)] = pack2(a0, a1);
    }
#pragma unroll
    for (int e = 0; e < 4; ++e) {
        int pidx = e * 256 + tid;
        int n = pidx >> 4, kp = (pidx & 15) * 2;
        const float* row = Wih + (size_t)(n0 + n) * 25;
        float b0 = row[kp     < 25 ? kp     : 24];
        float b1 = row[kp + 1 < 25 ? kp + 1 : 24];
        b0 = (kp     < 25) ? b0 : 0.f;
        b1 = (kp + 1 < 25) ? b1 : 0.f;
        ((unsigned int*)Bs)[n * (KT / 2) + (kp >> 1)] = pack2(b0, b1);
    }
    __syncthreads();
    v8f acc0 = {0.f,0.f,0.f,0.f,0.f,0.f,0.f,0.f};
    v8f acc1 = {0.f,0.f,0.f,0.f,0.f,0.f,0.f,0.f};
    v16bf af  = load_frag(As, wm * 16 + (lane & 15), g);
    v16bf bf0 = load_frag(Bs, wn * 32 + (lane & 15), g);
    v16bf bf1 = load_frag(Bs, wn * 32 + 16 + (lane & 15), g);
    acc0 = WMMA_BF16(af, bf0, acc0);
    acc1 = WMMA_BF16(af, bf1, acc1);

    int nA = n0 + wn * 32 + (lane & 15);
    int nB = nA + 16;
    float biasA = bih[nA] + bhh[nA];
    float biasB = bih[nB] + bhh[nB];
#pragma unroll
    for (int r = 0; r < 8; ++r) {
        int m = m0 + wm * 16 + r + 8 * g;
        Xg[(size_t)m * G4 + nA] = acc0[r] + biasA;
        Xg[(size_t)m * G4 + nB] = acc1[r] + biasB;
    }
}

// ---------------------------------------------------------------------------
// LSTM scan: one persistent 512-thread workgroup. W_hh (512x128 fp32) lives
// in LDS (rows padded to 129 floats -> conflict-free stride-128 row access,
// only possible with CDNA5's 320 KB/WGP LDS). Thread j owns gate element j;
// threads 0..127 own the cell state.
// ---------------------------------------------------------------------------
#define WROW 129
__global__ __launch_bounds__(512)
void lstm_kernel(const float* __restrict__ Xg, const float* __restrict__ Whh,
                 float* __restrict__ out) {
    extern __shared__ float smem[];
    float* wl  = smem;               // G4 * WROW
    float* h_s = wl + G4 * WROW;     // H_RNN
    float* g_s = h_s + H_RNN;        // G4

    const int tid = threadIdx.x;
    for (int i = tid; i < G4 * H_RNN; i += 512) {
        int j = i >> 7, k = i & 127;
        wl[j * WROW + k] = Whh[i];
    }
    if (tid < H_RNN) h_s[tid] = 0.f;
    float c = 0.f;
    __syncthreads();

    const float* wrow = wl + tid * WROW;
    for (int t = 0; t < M_TOTAL; ++t) {
        float dot = 0.f;
#pragma unroll 8
        for (int k = 0; k < H_RNN; ++k)
            dot = __builtin_fmaf(wrow[k], h_s[k], dot);   // h_s is a broadcast read
        g_s[tid] = Xg[(size_t)t * G4 + tid] + dot;
        __syncthreads();
        if (tid < H_RNN) {
            float iv = 1.f / (1.f + __expf(-g_s[tid]));
            float fv = 1.f / (1.f + __expf(-g_s[H_RNN + tid]));
            float gv = tanhf(g_s[2 * H_RNN + tid]);
            float ov = 1.f / (1.f + __expf(-g_s[3 * H_RNN + tid]));
            c = fv * c + iv * gv;
            float h = ov * tanhf(c);
            h_s[tid] = h;
            float y = h > 0.f ? h : 0.f;
            out[(size_t)t * OUT_COLS + D_FNN + tid] = y;
            out[(size_t)OUT_TOT + (size_t)t * OUT_COLS + D_FNN + tid] = y;
        }
        __syncthreads();
    }
}

// ---------------------------------------------------------------------------
extern "C" void kernel_launch(void* const* d_in, const int* in_sizes, int n_in,
                              void* d_out, int out_size, void* d_ws, size_t ws_size,
                              hipStream_t stream) {
    const float* features = (const float*)d_in[0];
    const float* W1  = (const float*)d_in[1];
    const float* b1  = (const float*)d_in[2];
    const float* W2  = (const float*)d_in[3];
    const float* b2  = (const float*)d_in[4];
    const float* Wih = (const float*)d_in[5];
    const float* bih = (const float*)d_in[6];
    const float* Whh = (const float*)d_in[7];
    const float* bhh = (const float*)d_in[8];
    float* out = (float*)d_out;

    // workspace: [bf16 hidden 32 MB][fp32 x_gates 64 MB]
    unsigned short* Hbf = (unsigned short*)d_ws;
    float* Xg = (float*)((char*)d_ws + (size_t)M_TOTAL * H_FNN * sizeof(unsigned short));

    dim3 blk(256);
    dim3 g1(H_FNN / NTW, M_TOTAL / MT);   // 8 x 512
    dim3 g3(G4 / NTW,    M_TOTAL / MT);   // 8 x 512
    dim3 g2(D_FNN / NTW, M_TOTAL / MT);   // 4 x 512

    hipLaunchKernelGGL(gemm1_kernel, g1, blk, 0, stream, features, W1, b1, Hbf);
    hipLaunchKernelGGL(gemm3_kernel, g3, blk, 0, stream, features, Wih, bih, bhh, Xg);
    hipLaunchKernelGGL(gemm2_kernel, g2, blk, 0, stream, Hbf, W2, b2, out);

    size_t lstm_lds = (size_t)(G4 * WROW + H_RNN + G4) * sizeof(float);  // ~260.5 KB (<320 KB/WGP)
    (void)hipFuncSetAttribute((const void*)lstm_kernel,
                              hipFuncAttributeMaxDynamicSharedMemorySize,
                              (int)lstm_lds);
    hipLaunchKernelGGL(lstm_kernel, dim3(1), dim3(512), lstm_lds, stream, Xg, Whh, out);
}
